// MultiHeadSelfAttention_52441550684709
// MI455X (gfx1250) — matrix operand order repaired and checked by reading, not compiled, not run on hardware
//
#include <hip/hip_runtime.h>
#include <hip/hip_bf16.h>

typedef __attribute__((ext_vector_type(16))) _Float16 v16h;
typedef __attribute__((ext_vector_type(8)))  _Float16 v8h;
typedef __attribute__((ext_vector_type(8)))  float    v8f;

constexpr int BATCH = 4;
constexpr int SEQ   = 2048;
constexpr int CMODEL= 1024;
constexpr int HN    = 16;
constexpr int DH    = 64;
constexpr int MTOT  = BATCH * SEQ;   // 8192 rows

#define WMMA16(a, b, c) \
    __builtin_amdgcn_wmma_f32_16x16x32_f16(false, (a), false, (b), (short)0, (c), false, false)

static __device__ __forceinline__ v16h cat8(v8h lo, v8h hi) {
    v16h r;
#pragma unroll
    for (int i = 0; i < 8; ++i) { r[i] = lo[i]; r[i + 8] = hi[i]; }
    return r;
}

// ---------------------------------------------------------------- converts
__global__ __launch_bounds__(256) void f32_to_f16_kernel(const float* __restrict__ in,
                                                         _Float16* __restrict__ out, int n) {
    for (int i = blockIdx.x * blockDim.x + threadIdx.x; i < n; i += gridDim.x * blockDim.x)
        out[i] = (_Float16)in[i];
}

// ---------------------------------------------------------------- GEMM: Y = A @ W^T + bias
// 32x64 tile per wave: 2 A fragments x 4 B fragments -> 8 WMMAs per 32-deep K step.
// A: [M, K] f16 row-major.  W: [N, K] f16 row-major (B-operand rows K-contiguous).
// MODE 1: f16 out, [B, H, T, Dh] head-split layout (Q, K)
// MODE 2: f16 out, [B, H, Dh, T] transposed layout (V)
// MODE 3: f32 out, [M, N] row-major (final projection -> d_out)
template <int MODE>
__global__ __launch_bounds__(256) void gemm_wmma_kernel(const _Float16* __restrict__ A,
                                                        const _Float16* __restrict__ W,
                                                        const float* __restrict__ bias,
                                                        _Float16* __restrict__ outh,
                                                        float* __restrict__ outf) {
    constexpr int K = CMODEL;
    const int lane = threadIdx.x & 31;
    const int wave = threadIdx.x >> 5;
    const int half = lane >> 4;
    const int lm   = lane & 15;

    const int nt = blockIdx.x;                 // N/64 tiles (16)
    const int mt = blockIdx.y * 8 + wave;      // M/32 tiles (256)
    const int m0 = mt * 32, n0 = nt * 64;

    const _Float16* __restrict__ arow0 = A + (size_t)(m0 + lm) * K;
    const _Float16* __restrict__ arow1 = A + (size_t)(m0 + 16 + lm) * K;
    const _Float16* __restrict__ brow0 = W + (size_t)(n0 + lm) * K;
    const _Float16* __restrict__ brow1 = W + (size_t)(n0 + 16 + lm) * K;
    const _Float16* __restrict__ brow2 = W + (size_t)(n0 + 32 + lm) * K;
    const _Float16* __restrict__ brow3 = W + (size_t)(n0 + 48 + lm) * K;

    v8f acc[2][4] = {};
#pragma unroll 2
    for (int kk = 0; kk < K; kk += 32) {
        const int ka = kk + 16 * half;
        const int kb = kk + 16 * half;
        const v16h a0 = cat8(*(const v8h*)(arow0 + ka), *(const v8h*)(arow0 + ka + 8));
        const v16h a1 = cat8(*(const v8h*)(arow1 + ka), *(const v8h*)(arow1 + ka + 8));

        const _Float16* br[4] = {brow0, brow1, brow2, brow3};
#pragma unroll
        for (int j = 0; j < 4; ++j) {
            const v16h b = cat8(*(const v8h*)(br[j] + kb), *(const v8h*)(br[j] + kb + 8));
            acc[0][j] = WMMA16(a0, b, acc[0][j]);
            acc[1][j] = WMMA16(a1, b, acc[1][j]);
        }
    }

#pragma unroll
    for (int j = 0; j < 4; ++j) {
        const int n  = n0 + j * 16 + lm;
        const float bvj = bias[n];
#pragma unroll
        for (int mi = 0; mi < 2; ++mi) {
#pragma unroll
            for (int r = 0; r < 8; ++r) {
                const int m = m0 + mi * 16 + r + 8 * half;
                const float v = acc[mi][j][r] + bvj;
                if constexpr (MODE == 1) {
                    const int bb = m >> 11, t = m & (SEQ - 1);
                    const int hh = n >> 6,  dd = n & (DH - 1);
                    outh[(((size_t)(bb * HN + hh)) * SEQ + t) * DH + dd] = (_Float16)v;
                } else if constexpr (MODE == 2) {
                    const int bb = m >> 11, t = m & (SEQ - 1);
                    const int hh = n >> 6,  dd = n & (DH - 1);
                    outh[(((size_t)(bb * HN + hh)) * DH + dd) * SEQ + t] = (_Float16)v;
                } else {
                    outf[(size_t)m * CMODEL + n] = v;
                }
            }
        }
    }
}

// ---------------------------------------------------------------- flash attention
// Q,K: [B,H,T,Dh] f16.  Vt: [B,H,Dh,T] f16.  Out: [B,T,H*Dh] f16.
// One wave32 per 16-row query tile; online softmax in exp2 domain; causal mask.
__global__ __launch_bounds__(256) void attn_wmma_kernel(const _Float16* __restrict__ Q,
                                                        const _Float16* __restrict__ Kb,
                                                        const _Float16* __restrict__ Vt,
                                                        _Float16* __restrict__ Aout) {
    __shared__ _Float16 pstage[8][16 * 32];   // per-wave P re-layout buffer (1 KB each)

    const int lane = threadIdx.x & 31;
    const int wave = threadIdx.x >> 5;
    const int half = lane >> 4;
    const int lm   = lane & 15;

    const int qt = blockIdx.x * 8 + wave;     // 0..127 query tiles
    const int h  = blockIdx.y;
    const int b  = blockIdx.z;
    const int bh = b * HN + h;
    const int q0 = qt * 16;

    const _Float16* __restrict__ Qh = Q  + (size_t)bh * SEQ * DH;
    const _Float16* __restrict__ Kh = Kb + (size_t)bh * SEQ * DH;
    const _Float16* __restrict__ Vh = Vt + (size_t)bh * DH * SEQ;   // [d][t]

    // Preload the Q tile as two A-operands (d-chunks 0 and 32).
    v16h aq[2];
    {
        const _Float16* qrow = Qh + (size_t)(q0 + lm) * DH;
#pragma unroll
        for (int c = 0; c < 2; ++c) {
            const int kk = c * 32;
            v8h a0 = *(const v8h*)(qrow + kk + 16 * half);
            v8h a1 = *(const v8h*)(qrow + kk + 16 * half + 8);
            aq[c] = cat8(a0, a1);
        }
    }

    float mrow[8], lrow[8];
    v8f o[4] = {{}, {}, {}, {}};
#pragma unroll
    for (int r = 0; r < 8; ++r) { mrow[r] = -1e30f; lrow[r] = 0.0f; }

    // logits scaled once by (1/sqrt(Dh)) * log2(e); softmax done with exp2 (native v_exp_f32)
    const float scale2 = 0.125f * 1.44269504088896340736f;
    _Float16* pbuf = &pstage[wave][0];

    for (int kc = 0; kc < q0 + 16; kc += 32) {
        // ---- S = Q K^T for keys [kc, kc+31] : two 16x16 tiles
        v8f s0 = {}, s1 = {};
#pragma unroll
        for (int c = 0; c < 2; ++c) {
            const int dk = c * 32;
            const _Float16* k0row = Kh + (size_t)(kc + lm) * DH + dk;
            const _Float16* k1row = Kh + (size_t)(kc + 16 + lm) * DH + dk;
            v8h t0 = *(const v8h*)(k0row + 16 * half);
            v8h t1 = *(const v8h*)(k0row + 16 * half + 8);
            v8h u0 = *(const v8h*)(k1row + 16 * half);
            v8h u1 = *(const v8h*)(k1row + 16 * half + 8);
            s0 = WMMA16(aq[c], cat8(t0, t1), s0);
            s1 = WMMA16(aq[c], cat8(u0, u1), s1);
        }

        // ---- scale + causal mask + online softmax (row = r + 8*half, col = lm)
#pragma unroll
        for (int r = 0; r < 8; ++r) {
            const int qi  = q0 + r + 8 * half;
            const int k0i = kc + lm;
            const int k1i = kc + 16 + lm;
            float v0 = s0[r] * scale2;
            float v1 = s1[r] * scale2;
            if (k0i > qi) v0 = -1e30f;
            if (k1i > qi) v1 = -1e30f;

            float mx = fmaxf(v0, v1);
#pragma unroll
            for (int off = 8; off; off >>= 1) mx = fmaxf(mx, __shfl_xor(mx, off, 16));
            const float mnew = fmaxf(mrow[r], mx);
            const float corr = exp2f(mrow[r] - mnew);
            const float p0 = exp2f(v0 - mnew);
            const float p1 = exp2f(v1 - mnew);
            float ps = p0 + p1;
#pragma unroll
            for (int off = 8; off; off >>= 1) ps += __shfl_xor(ps, off, 16);
            lrow[r] = lrow[r] * corr + ps;
            mrow[r] = mnew;
            o[0][r] *= corr; o[1][r] *= corr; o[2][r] *= corr; o[3][r] *= corr;
            s0[r] = p0; s1[r] = p1;
        }

        // ---- re-layout P (C-layout -> A-layout) through wave-private LDS
#pragma unroll
        for (int r = 0; r < 8; ++r) {
            pbuf[(r + 8 * half) * 32 + lm]      = (_Float16)s0[r];
            pbuf[(r + 8 * half) * 32 + 16 + lm] = (_Float16)s1[r];
        }
        const _Float16* prow = pbuf + lm * 32;
        v8h pl = *(const v8h*)(prow + 16 * half);
        v8h ph = *(const v8h*)(prow + 16 * half + 8);
        const v16h ap = cat8(pl, ph);

        // ---- O += P @ V : Vt rows are j-contiguous -> direct v16h B-operands
#pragma unroll
        for (int dt = 0; dt < 4; ++dt) {
            const v16h bv = *(const v16h*)(Vh + (size_t)(dt * 16 + lm) * SEQ + kc + 16 * half);
            o[dt] = WMMA16(ap, bv, o[dt]);
        }
    }

    // ---- epilogue: normalize by row sum, store [B, T, H*Dh] f16
#pragma unroll
    for (int r = 0; r < 8; ++r) {
        const float inv = 1.0f / lrow[r];
        const int row = q0 + r + 8 * half;
        _Float16* orow = Aout + ((size_t)(b * SEQ + row)) * CMODEL + h * DH;
#pragma unroll
        for (int dt = 0; dt < 4; ++dt)
            orow[dt * 16 + lm] = (_Float16)(o[dt][r] * inv);
    }
}

// ---------------------------------------------------------------- launch
extern "C" void kernel_launch(void* const* d_in, const int* in_sizes, int n_in,
                              void* d_out, int out_size, void* d_ws, size_t ws_size,
                              hipStream_t stream) {
    const float* x  = (const float*)d_in[0];
    const float* Wq = (const float*)d_in[1];
    const float* bq = (const float*)d_in[2];
    const float* Wk = (const float*)d_in[3];
    const float* bk = (const float*)d_in[4];
    const float* Wv = (const float*)d_in[5];
    const float* bv = (const float*)d_in[6];
    const float* Wo = (const float*)d_in[7];
    const float* bo = (const float*)d_in[8];
    float* out = (float*)d_out;

    char* ws = (char*)d_ws;
    const size_t MB = 1u << 20;
    _Float16* xh  = (_Float16*)(ws + 0 * MB);   // 16 MB
    _Float16* wqh = (_Float16*)(ws + 16 * MB);  //  2 MB
    _Float16* wkh = (_Float16*)(ws + 18 * MB);
    _Float16* wvh = (_Float16*)(ws + 20 * MB);
    _Float16* woh = (_Float16*)(ws + 22 * MB);
    _Float16* qb  = (_Float16*)(ws + 24 * MB);  // 16 MB [B,H,T,Dh]
    _Float16* kb  = (_Float16*)(ws + 40 * MB);  // 16 MB [B,H,T,Dh]
    _Float16* vtb = (_Float16*)(ws + 56 * MB);  // 16 MB [B,H,Dh,T]
    _Float16* ao  = (_Float16*)(ws + 72 * MB);  // 16 MB [B,T,C]

    // f32 -> f16 converts (x + 4 weight matrices)
    f32_to_f16_kernel<<<4096, 256, 0, stream>>>(x,  xh,  MTOT * CMODEL);
    f32_to_f16_kernel<<<1024, 256, 0, stream>>>(Wq, wqh, CMODEL * CMODEL);
    f32_to_f16_kernel<<<1024, 256, 0, stream>>>(Wk, wkh, CMODEL * CMODEL);
    f32_to_f16_kernel<<<1024, 256, 0, stream>>>(Wv, wvh, CMODEL * CMODEL);
    f32_to_f16_kernel<<<1024, 256, 0, stream>>>(Wo, woh, CMODEL * CMODEL);

    // Q/K/V projections (WMMA, 32x64 wave tiles); V written transposed.
    dim3 ggrid(CMODEL / 64, MTOT / (32 * 8));   // 16 x 32
    gemm_wmma_kernel<1><<<ggrid, 256, 0, stream>>>(xh, wqh, bq, qb,  nullptr);
    gemm_wmma_kernel<1><<<ggrid, 256, 0, stream>>>(xh, wkh, bk, kb,  nullptr);
    gemm_wmma_kernel<2><<<ggrid, 256, 0, stream>>>(xh, wvh, bv, vtb, nullptr);

    // Flash attention: one wave per 16-row query tile.
    dim3 agrid((SEQ / 16) / 8, HN, BATCH);      // 16 x 16 x 4
    attn_wmma_kernel<<<agrid, 256, 0, stream>>>(qb, kb, vtb, ao);

    // Output projection -> f32 d_out.
    gemm_wmma_kernel<3><<<ggrid, 256, 0, stream>>>(ao, woh, bo, nullptr, out);
}
